// Attention_13572096655603
// MI455X (gfx1250) — compile-verified, matrix operand order checked by reading, not run on hardware
//
#include <hip/hip_runtime.h>
#include <hip/hip_bf16.h>
#include <math.h>

// ---------------- CDNA5 WMMA types & fragment helpers ----------------
typedef __attribute__((ext_vector_type(16))) _Float16 v16h;
typedef __attribute__((ext_vector_type(8)))  float    v8f;
typedef __attribute__((ext_vector_type(8)))  _Float16 half8;

// Assemble A/B fragment (16x32 f16, K-packed per lane) from row-major f16 with
// leading dim `ld` (multiple of 8 -> 16B aligned vector loads).
// lane<16 : row rc, k = {k0..k0+7, k0+16..k0+23}; lane>=16: k shifted by +8.
static __device__ __forceinline__ v16h load_frag(const _Float16* __restrict__ p,
                                                 int ld, int rc, int k0, int lane) {
  const _Float16* base = p + (size_t)rc * ld + k0 + (lane >> 4) * 8;
  half8 lo = *(const half8*)(base);
  half8 hi = *(const half8*)(base + 16);
  v16h a;
#pragma unroll
  for (int e = 0; e < 8; ++e) { a[e] = lo[e]; a[e + 8] = hi[e]; }
  return a;
}

static __device__ __forceinline__ v8f wmma_f16(v16h a, v16h b, v8f c) {
  return __builtin_amdgcn_wmma_f32_16x16x32_f16(false, a, false, b, (short)0, c,
                                                false, false);
}

static __device__ __forceinline__ float sigmf(float x) {
  return 1.0f / (1.0f + __expf(-x));
}

#define BB   32
#define NN   512
#define DD   256
#define HH   256
#define G4   1024
#define TT   2048

// ---------------- conversion kernels ----------------
__global__ void cvt_f16_kernel(const float* __restrict__ in,
                               _Float16* __restrict__ out, int n) {
  int i = blockIdx.x * blockDim.x + threadIdx.x;
  if (i < n) out[i] = (_Float16)in[i];
}

// rng W is [1024, 257]: split into main [1024,256] f16 (ld=256) + extra col f32
__global__ void cvt_w257_kernel(const float* __restrict__ in,
                                _Float16* __restrict__ wmain,
                                float* __restrict__ wextra) {
  int c = blockIdx.x;
  for (int k = threadIdx.x; k < 256; k += blockDim.x)
    wmain[c * 256 + k] = (_Float16)in[c * 257 + k];
  if (threadIdx.x == 0) wextra[c] = in[c * 257 + 256];
}

// fused convert + transpose: emb [B,N,D] f32 -> xT [B,D,N] f16 (tiled, padded)
__global__ __launch_bounds__(256) void transpose_x_kernel(
    const float* __restrict__ emb, _Float16* __restrict__ xT) {
  __shared__ _Float16 tile[32][33];
  const int b = blockIdx.z;
  const int n0 = blockIdx.x * 32, d0 = blockIdx.y * 32;
  const int tx = threadIdx.x, ty = threadIdx.y;  // (32, 8)
  for (int i = ty; i < 32; i += 8)
    tile[i][tx] = (_Float16)emb[((size_t)b * NN + (n0 + i)) * DD + d0 + tx];
  __syncthreads();
  for (int i = ty; i < 32; i += 8)
    xT[((size_t)b * DD + (d0 + i)) * NN + n0 + tx] = tile[tx][i];
}

// ---------------- input-gate GEMM: out[r,c] = X[r,:]·W[c,:] + bias[c] (+extra)
__global__ __launch_bounds__(256) void xgate_gemm_kernel(
    const _Float16* __restrict__ X, const _Float16* __restrict__ W,
    const float* __restrict__ bias, const float* __restrict__ extra_in,
    const float* __restrict__ extra_w, _Float16* __restrict__ out) {
  const int wave = threadIdx.x >> 5, lane = threadIdx.x & 31;
  const int tile = blockIdx.x * 8 + wave;
  const int ct = tile & 63, rt = tile >> 6;  // 64 col tiles, 1024 row tiles
  const int r0 = rt * 16, c0 = ct * 16;
  const int lr = lane & 15, hf = lane >> 4;
  v8f acc = {};
#pragma unroll
  for (int kk = 0; kk < 8; ++kk) {
    v16h a = load_frag(X, DD, r0 + lr, kk * 32, lane);
    v16h b = load_frag(W, DD, c0 + lr, kk * 32, lane);
    acc = wmma_f16(a, b, acc);
  }
  const int col = c0 + lr;
  const float bc = bias[col];
  const float ew = extra_w ? extra_w[col] : 0.0f;
#pragma unroll
  for (int r = 0; r < 8; ++r) {
    int row = r0 + r + 8 * hf;
    float v = acc[r] + bc;
    if (extra_in) v += extra_in[row] * ew;
    out[(size_t)row * G4 + col] = (_Float16)v;
  }
}

// ---------------- BiLSTM recurrence: one WGP-resident block per direction.
// 32 waves: wave w -> batch tile (w&1), hidden slice jt=(w>>1)*16.  Each wave
// computes the i,f,g,o tiles for its (batch,hidden) slice -> gate math fully
// in registers; c-state lives in the v8f accumulator layout across steps.
// Gates i,f of Whh (256 KB f16) are cached in LDS (CDNA5: 320 KB/WGP), gates
// g,o stream from L2 -> per-step off-WGP traffic halved on the critical path.
__global__ __launch_bounds__(1024) void lstm_kernel(
    const _Float16* __restrict__ xg_f, const _Float16* __restrict__ xg_b,
    const _Float16* __restrict__ whh_f, const _Float16* __restrict__ whh_b,
    _Float16* __restrict__ hs /* [B,N,2H] f16 */) {
  __shared__ _Float16 hsh[BB][HH];        // 16 KB current h (f16)
  __shared__ _Float16 whl[512 * HH];      // 256 KB: Whh rows 0..511 (gates i,f)
  const int dir = blockIdx.x;             // 0 fwd, 1 bwd
  const _Float16* __restrict__ xg  = dir ? xg_b : xg_f;
  const _Float16* __restrict__ whh = dir ? whh_b : whh_f;
  const int tid = threadIdx.x, wave = tid >> 5, lane = tid & 31;
  const int mt = wave & 1, jt = wave >> 1;
  const int lr = lane & 15, hf = lane >> 4;
  for (int i = tid; i < BB * HH; i += 1024) ((_Float16*)hsh)[i] = (_Float16)0.0f;
  {  // stage gates i,f of Whh into LDS (vectorized 16B copies)
    const half8* __restrict__ src = (const half8*)whh;
    half8* dst = (half8*)whl;
    for (int i = tid; i < (512 * HH) / 8; i += 1024) dst[i] = src[i];
  }
  v8f cst = {};  // c-state, D-matrix layout: row b = mt*16 + r + 8*hf, col lr
  __syncthreads();

  for (int step = 0; step < NN; ++step) {
    const int seq = dir ? (NN - 1 - step) : step;
    // prefetch next step's input-gate rows (global_prefetch_b8)
    if (step + 1 < NN) {
      const int seqn = dir ? (NN - 2 - step) : step + 1;
      __builtin_prefetch(
          &xg[(size_t)((mt * 16 + lr) * NN + seqn) * G4 + jt * 16], 0, 1);
    }
    // A fragments: h (f16) from LDS, row = batch
    v16h af[8];
    {
      const _Float16* hrow = &hsh[mt * 16 + lr][0];
#pragma unroll
      for (int kk = 0; kk < 8; ++kk) {
        const _Float16* base = hrow + kk * 32 + hf * 8;
        half8 lo = *(const half8*)(base);
        half8 hi = *(const half8*)(base + 16);
        v16h a;
#pragma unroll
        for (int e = 0; e < 8; ++e) { a[e] = lo[e]; a[e + 8] = hi[e]; }
        af[kk] = a;
      }
    }
    __syncthreads();  // all reads of h done before anyone writes new h

    float gv[4][8];
#pragma unroll
    for (int g = 0; g < 4; ++g) {
      const int c0 = g * HH + jt * 16;
      const int col = c0 + lr;
      v8f acc;
#pragma unroll
      for (int r = 0; r < 8; ++r) {  // init from precomputed input gates
        int b = mt * 16 + r + 8 * hf;
        acc[r] = (float)xg[(size_t)(b * NN + seq) * G4 + col];
      }
#pragma unroll
      for (int kk = 0; kk < 8; ++kk) {
        v16h bf;
        if (g < 2) bf = load_frag(whl, HH, col, kk * 32, lane);   // LDS
        else       bf = load_frag(whh, HH, col, kk * 32, lane);   // L2
        acc = wmma_f16(af[kk], bf, acc);
      }
#pragma unroll
      for (int r = 0; r < 8; ++r) gv[g][r] = acc[r];
    }

    const int hcol = jt * 16 + lr;
#pragma unroll
    for (int r = 0; r < 8; ++r) {
      float iv = sigmf(gv[0][r]), fv = sigmf(gv[1][r]);
      float gg = tanhf(gv[2][r]), ov = sigmf(gv[3][r]);
      float c = fv * cst[r] + iv * gg;
      cst[r] = c;
      _Float16 hh = (_Float16)(ov * tanhf(c));
      int b = mt * 16 + r + 8 * hf;
      hsh[b][hcol] = hh;
      hs[((size_t)b * NN + seq) * (2 * HH) + dir * HH + hcol] = hh;
    }
    __syncthreads();  // new h visible before next step's reads
  }
}

// ---------------- projection: out[row] = hs[row,:]·pw + pb (mode1 -> r^-2)
__global__ __launch_bounds__(256) void proj_kernel(
    const _Float16* __restrict__ hsb, const float* __restrict__ pw,
    const float* __restrict__ pb, float* __restrict__ out,
    float* __restrict__ out2, int mode) {
  int row = blockIdx.x * 8 + (threadIdx.x >> 5);
  int lane = threadIdx.x & 31;
  const _Float16* hr = hsb + (size_t)row * (2 * HH);
  float s = 0.0f;
  for (int j = lane; j < 2 * HH; j += 32) s += (float)hr[j] * pw[j];
#pragma unroll
  for (int off = 16; off; off >>= 1) s += __shfl_down(s, off, 32);
  if (lane == 0) {
    s += pb[0];
    if (mode == 1) out[row] = 1.0f / (s * s);          // ranges^-2
    else { out[row] = s; if (out2) out2[row] = s; }    // durations
  }
}

// ---------------- cumsum + centers (one block per batch) ----------------
__global__ __launch_bounds__(512) void cumsum_kernel(const float* __restrict__ dur,
                                                     float* __restrict__ centers) {
  __shared__ float buf[NN];
  int b = blockIdx.x, n = threadIdx.x;
  float d = dur[b * NN + n];
  buf[n] = d;
  __syncthreads();
  for (int off = 1; off < NN; off <<= 1) {
    float v = buf[n];
    float add = (n >= off) ? buf[n - off] : 0.0f;
    __syncthreads();
    buf[n] = v + add;
    __syncthreads();
  }
  centers[b * NN + n] = buf[n] - 0.5f * d;
}

// ---------------- denominator: rdenom[b,t] = 1/(sum_n exp(..) + eps) --------
__global__ __launch_bounds__(256) void denom_kernel(
    const float* __restrict__ centers, const float* __restrict__ alpha,
    float* __restrict__ rdenom) {
  __shared__ float cs[NN], as_[NN];
  int b = blockIdx.y;
  for (int i = threadIdx.x; i < NN; i += 256) {
    cs[i] = centers[b * NN + i];
    as_[i] = alpha[b * NN + i];
  }
  __syncthreads();
  int t = blockIdx.x * 256 + threadIdx.x;
  float tf = (float)t, s = 0.0f;
  for (int n = 0; n < NN; ++n) {
    float d = tf - cs[n];
    s += __expf(-as_[n] * d * d);
  }
  rdenom[b * TT + t] = 1.0f / (s + 1e-6f);
}

// ---------------- normalized weights: wbuf[b,t,n] f16 (coalesced in n) ------
__global__ __launch_bounds__(512) void wnorm_kernel(
    const float* __restrict__ centers, const float* __restrict__ alpha,
    const float* __restrict__ rdenom, _Float16* __restrict__ wbuf) {
  const int b = blockIdx.y, t = blockIdx.x, n = threadIdx.x;
  float d = (float)t - centers[b * NN + n];
  float w = __expf(-alpha[b * NN + n] * d * d) * rdenom[b * TT + t];
  wbuf[((size_t)b * TT + t) * NN + n] = (_Float16)w;
}

// ---------------- attention einsum as pure WMMA GEMM ------------------------
// att[b,t,d] = sum_n wbuf[b,t,n] * xT[b,d,n]  (both operands K-contiguous)
__global__ __launch_bounds__(256) void att_kernel(
    const _Float16* __restrict__ wbuf, const _Float16* __restrict__ xT,
    const float* __restrict__ pe, float* __restrict__ out) {
  const int b = blockIdx.z, tt = blockIdx.y, dgrp = blockIdx.x;
  const int wave = threadIdx.x >> 5, lane = threadIdx.x & 31;
  const int d0 = (dgrp * 8 + wave) * 16, t0 = tt * 16;
  const int lr = lane & 15, hf = lane >> 4;
  const _Float16* __restrict__ wb = wbuf + (size_t)b * TT * NN;
  const _Float16* __restrict__ xb = xT + (size_t)b * DD * NN;
  v8f acc = {};
#pragma unroll
  for (int kk = 0; kk < 16; ++kk) {
    v16h a  = load_frag(wb, NN, t0 + lr, kk * 32, lane);
    v16h bf = load_frag(xb, NN, d0 + lr, kk * 32, lane);
    acc = wmma_f16(a, bf, acc);
  }
#pragma unroll
  for (int r = 0; r < 8; ++r) {
    int t = t0 + r + 8 * hf;
    int d = d0 + lr;
    out[((size_t)b * TT + t) * DD + d] = acc[r] + pe[t * DD + d];
  }
}

// ---------------- launch ----------------
extern "C" void kernel_launch(void* const* d_in, const int* in_sizes, int n_in,
                              void* d_out, int out_size, void* d_ws, size_t ws_size,
                              hipStream_t stream) {
  (void)in_sizes; (void)n_in; (void)out_size; (void)ws_size;
  const float* emb    = (const float*)d_in[0];
  const float* dur_Wf = (const float*)d_in[3];
  const float* dur_Uf = (const float*)d_in[4];
  const float* dur_bf = (const float*)d_in[5];
  const float* dur_Wb = (const float*)d_in[6];
  const float* dur_Ub = (const float*)d_in[7];
  const float* dur_bb = (const float*)d_in[8];
  const float* dur_pw = (const float*)d_in[9];
  const float* dur_pb = (const float*)d_in[10];
  const float* rng_Wf = (const float*)d_in[11];
  const float* rng_Uf = (const float*)d_in[12];
  const float* rng_bf = (const float*)d_in[13];
  const float* rng_Wb = (const float*)d_in[14];
  const float* rng_Ub = (const float*)d_in[15];
  const float* rng_bb = (const float*)d_in[16];
  const float* rng_pw = (const float*)d_in[17];
  const float* rng_pb = (const float*)d_in[18];
  const float* pe     = (const float*)d_in[19];

  char* base = (char*)d_ws;
  size_t off = 0;
  auto take = [&](size_t bytes) {
    void* r = base + off;
    off = (off + bytes + 255) & ~(size_t)255;
    return r;
  };
  const size_t R = (size_t)BB * NN;  // 16384 rows
  _Float16* xh    = (_Float16*)take(R * DD * 2);
  _Float16* xT    = (_Float16*)take(R * DD * 2);
  _Float16* wdf16 = (_Float16*)take(G4 * DD * 2);
  _Float16* udf16 = (_Float16*)take(G4 * HH * 2);
  _Float16* wdb16 = (_Float16*)take(G4 * DD * 2);
  _Float16* udb16 = (_Float16*)take(G4 * HH * 2);
  _Float16* wrf16 = (_Float16*)take(G4 * DD * 2);
  _Float16* urf16 = (_Float16*)take(G4 * HH * 2);
  _Float16* wrb16 = (_Float16*)take(G4 * DD * 2);
  _Float16* urb16 = (_Float16*)take(G4 * HH * 2);
  float* ewf      = (float*)take(G4 * 4);
  float* ewb      = (float*)take(G4 * 4);
  _Float16* xg_df = (_Float16*)take(R * G4 * 2);
  _Float16* xg_db = (_Float16*)take(R * G4 * 2);
  _Float16* xg_rf = (_Float16*)take(R * G4 * 2);
  _Float16* xg_rb = (_Float16*)take(R * G4 * 2);
  _Float16* hd    = (_Float16*)take(R * 2 * HH * 2);
  _Float16* hr    = (_Float16*)take(R * 2 * HH * 2);
  _Float16* wbuf  = (_Float16*)take((size_t)BB * TT * NN * 2);
  float* durs     = (float*)take(R * 4);
  float* alpha    = (float*)take(R * 4);
  float* centers  = (float*)take(R * 4);
  float* rdenom   = (float*)take((size_t)BB * TT * 4);

  const int EW = G4 * DD;  // 262144 weight elements
  // f32 -> f16 conversions + transposed x panel
  cvt_f16_kernel<<<(int)((R * DD + 255) / 256), 256, 0, stream>>>(emb, xh, (int)(R * DD));
  transpose_x_kernel<<<dim3(NN / 32, DD / 32, BB), dim3(32, 8), 0, stream>>>(emb, xT);
  cvt_f16_kernel<<<(EW + 255) / 256, 256, 0, stream>>>(dur_Wf, wdf16, EW);
  cvt_f16_kernel<<<(EW + 255) / 256, 256, 0, stream>>>(dur_Uf, udf16, EW);
  cvt_f16_kernel<<<(EW + 255) / 256, 256, 0, stream>>>(dur_Wb, wdb16, EW);
  cvt_f16_kernel<<<(EW + 255) / 256, 256, 0, stream>>>(dur_Ub, udb16, EW);
  cvt_f16_kernel<<<(EW + 255) / 256, 256, 0, stream>>>(rng_Uf, urf16, EW);
  cvt_f16_kernel<<<(EW + 255) / 256, 256, 0, stream>>>(rng_Ub, urb16, EW);
  cvt_w257_kernel<<<G4, 256, 0, stream>>>(rng_Wf, wrf16, ewf);
  cvt_w257_kernel<<<G4, 256, 0, stream>>>(rng_Wb, wrb16, ewb);

  const int GEMM_BLOCKS = (int)(R / 16) * 64 / 8;  // 8192
  // duration predictor
  xgate_gemm_kernel<<<GEMM_BLOCKS, 256, 0, stream>>>(xh, wdf16, dur_bf, nullptr, nullptr, xg_df);
  xgate_gemm_kernel<<<GEMM_BLOCKS, 256, 0, stream>>>(xh, wdb16, dur_bb, nullptr, nullptr, xg_db);
  lstm_kernel<<<2, 1024, 0, stream>>>(xg_df, xg_db, udf16, udb16, hd);
  proj_kernel<<<(int)(R / 8), 256, 0, stream>>>(hd, dur_pw, dur_pb, durs, (float*)d_out, 0);
  cumsum_kernel<<<BB, NN, 0, stream>>>(durs, centers);
  // range predictor (input = [x | durations])
  xgate_gemm_kernel<<<GEMM_BLOCKS, 256, 0, stream>>>(xh, wrf16, rng_bf, durs, ewf, xg_rf);
  xgate_gemm_kernel<<<GEMM_BLOCKS, 256, 0, stream>>>(xh, wrb16, rng_bb, durs, ewb, xg_rb);
  lstm_kernel<<<2, 1024, 0, stream>>>(xg_rf, xg_rb, urf16, urb16, hr);
  proj_kernel<<<(int)(R / 8), 256, 0, stream>>>(hr, rng_pw, rng_pb, alpha, nullptr, 1);
  // Gaussian upsampling
  denom_kernel<<<dim3(TT / 256, BB), 256, 0, stream>>>(centers, alpha, rdenom);
  wnorm_kernel<<<dim3(TT, BB), NN, 0, stream>>>(centers, alpha, rdenom, wbuf);
  att_kernel<<<dim3(2, TT / 16, BB), 256, 0, stream>>>(wbuf, xT, pe, (float*)d_out + R);
}
